// TransformerEncoder_29386166239652
// MI455X (gfx1250) — matrix operand order repaired and checked by reading, not compiled, not run on hardware
//
#include <hip/hip_runtime.h>

// ---------------------------------------------------------------------------
// Problem constants (match reference)
// ---------------------------------------------------------------------------
constexpr int BB = 4;        // batch
constexpr int SS = 2048;     // seq len
constexpr int DD = 512;      // embed dim
constexpr int HH = 4;        // heads (each full-dim)
constexpr int FF = 2048;     // mlp hidden
constexpr int MM = BB * SS;  // 8192 token rows

// ---------------------------------------------------------------------------
// Types for WMMA bf16 (gfx1250, wave32)
// ---------------------------------------------------------------------------
typedef __attribute__((ext_vector_type(16))) __bf16 v16bf;
typedef __attribute__((ext_vector_type(8)))  float  v8f;
typedef __attribute__((ext_vector_type(4)))  unsigned u32x4;
typedef __attribute__((ext_vector_type(2)))  unsigned u32x2;
typedef __attribute__((ext_vector_type(8)))  int      i32x8;
typedef __attribute__((ext_vector_type(4)))  int      i32x4;

// Tensor Data Mover availability (this toolchain: 6-arg clang-23 form)
#if defined(__has_builtin)
#if __has_builtin(__builtin_amdgcn_tensor_load_to_lds) && \
    __has_builtin(__builtin_amdgcn_s_wait_tensorcnt)
#define USE_TDM 1
#endif
#endif
#ifndef USE_TDM
#define USE_TDM 0
#endif

union FragBF {
    v16bf v;
    unsigned u[8];
    u32x4 q[2];
    u32x2 d2[4];
};
static_assert(sizeof(FragBF) == 32, "frag size");

__device__ __forceinline__ unsigned short f2bf(float f) {
    unsigned u = __float_as_uint(f);
    u += 0x7fffu + ((u >> 16) & 1u);   // round-to-nearest-even
    return (unsigned short)(u >> 16);
}

// ---------------------------------------------------------------------------
// fp32 -> bf16 elementwise convert (weights prep)
// ---------------------------------------------------------------------------
__global__ __launch_bounds__(256) void f32_to_bf16_kernel(
    const float* __restrict__ in, unsigned short* __restrict__ out, long long n) {
    long long i = (long long)blockIdx.x * 256 + threadIdx.x;
    if (i < n) out[i] = f2bf(in[i]);
}

// ---------------------------------------------------------------------------
// LayerNorm over D=512, one block (256 thr) per row, bf16 output
// ---------------------------------------------------------------------------
__global__ __launch_bounds__(256) void layernorm_bf16_kernel(
    const float* __restrict__ x, const float* __restrict__ gamma,
    const float* __restrict__ beta, unsigned short* __restrict__ out) {
    const int D = DD;
    int row = blockIdx.x;
    const float* xr = x + (size_t)row * D;
    int t = threadIdx.x;
    float v0 = xr[t], v1 = xr[t + 256];
    float s = v0 + v1, sq = v0 * v0 + v1 * v1;
#pragma unroll
    for (int mk = 16; mk >= 1; mk >>= 1) {
        s  += __shfl_xor(s,  mk, 32);
        sq += __shfl_xor(sq, mk, 32);
    }
    __shared__ float ss[8], ssq[8];
    int wave = t >> 5, lane = t & 31;
    if (lane == 0) { ss[wave] = s; ssq[wave] = sq; }
    __syncthreads();
    float st = 0.f, sqt = 0.f;
#pragma unroll
    for (int w = 0; w < 8; ++w) { st += ss[w]; sqt += ssq[w]; }
    float mean = st * (1.0f / 512.0f);
    float var  = sqt * (1.0f / 512.0f) - mean * mean;
    float r = rsqrtf(var + 1e-5f);
    out[(size_t)row * D + t]       = f2bf((v0 - mean) * r * gamma[t]       + beta[t]);
    out[(size_t)row * D + t + 256] = f2bf((v1 - mean) * r * gamma[t + 256] + beta[t + 256]);
}

// ---------------------------------------------------------------------------
// Mean over H=4 heads, bf16 output
// ---------------------------------------------------------------------------
__global__ __launch_bounds__(256) void avg_heads_kernel(
    const float* __restrict__ O, unsigned short* __restrict__ avg, long long md) {
    long long i = (long long)blockIdx.x * 256 + threadIdx.x;
    if (i < md) {
        float s = O[i] + O[md + i] + O[2 * md + i] + O[3 * md + i];
        avg[i] = f2bf(0.25f * s);
    }
}

// ---------------------------------------------------------------------------
// Tiled WMMA bf16 GEMM: C[M,N] = A[M,K] * B[K,N] + epilogue
//   block = 128 threads = 4 waves (2x2), block tile 128x128, K step 32
//   EPI 0: bf16 C = acc + bias            (QKV projections)
//   EPI 1: f32  C = acc + bias + R        (Wo / W2 with residual)
//   EPI 2: bf16 C = gelu(acc + bias)      (W1)
// ---------------------------------------------------------------------------
constexpr int AS_STRIDE = 20;  // dwords per 32-bf16 A row (16 + 4 pad, 16B aligned)
constexpr int BS_STRIDE = 20;  // dwords per 32-bf16 B^T row

template <int EPI>
__global__ __launch_bounds__(128) void gemm_bf16_kernel(
    const unsigned short* __restrict__ A,
    const unsigned short* __restrict__ Bw,
    const float* __restrict__ bias,
    const float* __restrict__ R,
    unsigned short* __restrict__ Cb,
    float* __restrict__ Cf,
    int M, int N, int K,
    long long bStrideZ, long long biasStrideZ, long long cStrideZ) {
    __shared__ unsigned As[128 * AS_STRIDE];
    __shared__ unsigned Bs[128 * BS_STRIDE];

    int z = blockIdx.z;
    Bw   += (size_t)z * bStrideZ;
    bias += (size_t)z * biasStrideZ;
    if (EPI == 0 || EPI == 2) Cb += (size_t)z * cStrideZ;

    int n0 = blockIdx.x * 128;
    int m0 = blockIdx.y * 128;
    int tid = threadIdx.x;
    int lane = tid & 31, wave = tid >> 5;
    int wm = (wave >> 1) * 64, wn = (wave & 1) * 64;
    int nn = lane & 15, kh = lane >> 4;

    v8f c[4][4];
    const v8f vzero = {0.f, 0.f, 0.f, 0.f, 0.f, 0.f, 0.f, 0.f};
#pragma unroll
    for (int i = 0; i < 4; ++i)
#pragma unroll
        for (int j = 0; j < 4; ++j) c[i][j] = vzero;

    const unsigned* A32 = (const unsigned*)A;
    const int Kdw = K >> 1;
    unsigned short* Bs16 = (unsigned short*)Bs;

    for (int k0 = 0; k0 < K; k0 += 32) {
        // --- stage A tile (row-major): thread t loads row m0+t, 16 dwords ---
        {
            const u32x4* src = (const u32x4*)(A32 + (size_t)(m0 + tid) * Kdw + (k0 >> 1));
            u32x4* dst = (u32x4*)(&As[tid * AS_STRIDE]);
#pragma unroll
            for (int i = 0; i < 4; ++i) dst[i] = src[i];
        }
        // --- stage B tile transposed: Bs[n][k] = B[k0+k][n0+n] ---
#pragma unroll
        for (int i = 0; i < 8; ++i) {
            int idx = i * 512 + tid * 4;
            int kk = idx >> 7;
            int nc = idx & 127;
            u32x2 val = *(const u32x2*)(const void*)(Bw + (size_t)(k0 + kk) * N + n0 + nc);
            Bs16[(nc + 0) * (BS_STRIDE * 2) + kk] = (unsigned short)(val[0] & 0xffffu);
            Bs16[(nc + 1) * (BS_STRIDE * 2) + kk] = (unsigned short)(val[0] >> 16);
            Bs16[(nc + 2) * (BS_STRIDE * 2) + kk] = (unsigned short)(val[1] & 0xffffu);
            Bs16[(nc + 3) * (BS_STRIDE * 2) + kk] = (unsigned short)(val[1] >> 16);
        }
        // --- prefetch next K-step tiles into cache (global_prefetch_b8) ---
        if (k0 + 32 < K) {
            __builtin_prefetch(A32 + (size_t)(m0 + tid) * Kdw + ((k0 + 32) >> 1), 0, 1);
            __builtin_prefetch(Bw + (size_t)(k0 + 32 + (tid >> 2)) * N + n0 + (tid & 3) * 32,
                               0, 1);
        }
        __syncthreads();

        FragBF a[4], bf[4];
#pragma unroll
        for (int i = 0; i < 4; ++i) {
            int row = wm + i * 16 + nn;
            const u32x4* p = (const u32x4*)(&As[row * AS_STRIDE + kh * 4]);
            a[i].q[0] = p[0];
            a[i].q[1] = p[2];  // +8 dwords = K+16 half
        }
#pragma unroll
        for (int j = 0; j < 4; ++j) {
            int col = wn + j * 16 + nn;
            const u32x4* p = (const u32x4*)(&Bs[col * BS_STRIDE + kh * 4]);
            bf[j].q[0] = p[0];
            bf[j].q[1] = p[2];
        }
#pragma unroll
        for (int i = 0; i < 4; ++i)
#pragma unroll
            for (int j = 0; j < 4; ++j)
                c[i][j] = __builtin_amdgcn_wmma_f32_16x16x32_bf16(
                    false, a[i].v, false, bf[j].v, (short)0, c[i][j], false, false);
        __syncthreads();
    }

    // --- epilogue: C layout M = v + 8*kh, N = nn ---
#pragma unroll
    for (int i = 0; i < 4; ++i) {
#pragma unroll
        for (int j = 0; j < 4; ++j) {
            int colBase = n0 + wn + j * 16 + nn;
            float bcol = bias[colBase];
#pragma unroll
            for (int v = 0; v < 8; ++v) {
                int row = m0 + wm + i * 16 + v + 8 * kh;
                float acc = c[i][j][v] + bcol;
                size_t idx = (size_t)row * N + colBase;
                if constexpr (EPI == 0) {
                    Cb[idx] = f2bf(acc);
                } else if constexpr (EPI == 1) {
                    Cf[idx] = acc + R[idx];
                } else {
                    float ge = 0.5f * acc * (1.0f + erff(acc * 0.70710678118654752f));
                    Cb[idx] = f2bf(ge);
                }
            }
        }
    }
}

// ---------------------------------------------------------------------------
// Flash attention (one head, full D=512 per head). Grid: (S/64, H*B).
// Block = 128 threads = 4 waves; each wave owns 16 Q rows, accumulates full D.
// Dynamic LDS: K tile 64x512 bf16 (TDM-loaded, padded rows), V tile transposed
// 512x64 bf16, P 64x64 bf16.
// ---------------------------------------------------------------------------
constexpr int KS_STRIDE = 260;  // dwords per K-tile row (256 + 4 pad) == TDM pad 4dw/256dw
constexpr int VS_STRIDE = 33;   // dwords per V^T row (32 + 1 pad)
constexpr int PS_STRIDE = 34;   // dwords per P row (32 + 2 pad)
constexpr int FLASH_LDS_DW = 64 * KS_STRIDE + 512 * VS_STRIDE + 64 * PS_STRIDE;

__global__ __launch_bounds__(128) void flash_attn_kernel(
    const unsigned short* __restrict__ Q,
    const unsigned short* __restrict__ Kb,
    const unsigned short* __restrict__ Vb,
    float* __restrict__ O) {
    extern __shared__ unsigned smem[];
    unsigned* Ks = smem;                       // 64 * 260 dw
    unsigned* Vs = Ks + 64 * KS_STRIDE;        // 512 * 33 dw
    unsigned* Ps = Vs + 512 * VS_STRIDE;       // 64 * 34 dw
    unsigned short* Ps16 = (unsigned short*)Ps;
    unsigned short* Vs16 = (unsigned short*)Vs;

    int qblk = blockIdx.x;       // 0..31
    int hb   = blockIdx.y;       // 0..15
    int h = hb >> 2;             // / BB
    int b = hb & 3;
    int tid = threadIdx.x;
    int lane = tid & 31, wave = tid >> 5;
    int nn = lane & 15, kh = lane >> 4;
    size_t headBase = (size_t)h * MM + (size_t)b * SS;  // row offset into [H][M] rows
    int q0 = qblk * 64;

    const unsigned* Q32 = (const unsigned*)Q;
    const unsigned* V32 = (const unsigned*)Vb;
#if !USE_TDM
    const unsigned* K32 = (const unsigned*)Kb;
#endif

    // ---- preload Q A-fragments for this wave's 16 rows (16 K-steps of 32) ----
    FragBF qf[16];
    {
        size_t qrow = headBase + q0 + wave * 16 + nn;
        const unsigned* qp = Q32 + qrow * (DD / 2);
#pragma unroll
        for (int ds = 0; ds < 16; ++ds) {
            const u32x4* p = (const u32x4*)(qp + ds * 16 + kh * 4);
            qf[ds].q[0] = p[0];
            qf[ds].q[1] = p[2];
        }
    }

    v8f o[32];
    const v8f vzero = {0.f, 0.f, 0.f, 0.f, 0.f, 0.f, 0.f, 0.f};
#pragma unroll
    for (int n = 0; n < 32; ++n) o[n] = vzero;
    float mprev[8], lsum[8];
#pragma unroll
    for (int v = 0; v < 8; ++v) { mprev[v] = -1e30f; lsum[v] = 0.0f; }
    const float SCALE = 0.04419417382415922f;  // 1/sqrt(512)

    for (int t0 = 0; t0 < SS; t0 += 64) {
#if USE_TDM
        // ---- stage K tile via Tensor Data Mover: one 64x512 bf16 2D tile,
        //      LDS rows padded 4 dwords per 256 (matches KS_STRIDE) ----
        if (wave == 0) {
            unsigned ldsoff = (unsigned)(size_t)(void*)Ks;
            unsigned long long ga =
                (unsigned long long)(size_t)(Kb + (headBase + (size_t)t0) * DD);
            u32x4 g0;
            g0[0] = 1u;                                   // count=1, user mode
            g0[1] = ldsoff;                               // lds_addr (bytes)
            g0[2] = (unsigned)ga;                         // global_addr[31:0]
            g0[3] = (unsigned)((ga >> 32) & 0x01ffffffu)  // global_addr[56:32]
                    | 0x80000000u;                        // type=2 ("image")
            i32x8 g1;
            g1[0] = (int)((1u << 16)     // data_size = 2 bytes
                          | (1u << 20)   // pad_enable
                          | (7u << 22)   // pad_interval: 256 dwords
                          | (3u << 25)); // pad_amount: 4 dwords
            g1[1] = (int)(512u << 16);   // tensor_dim0 = 512 elements
            g1[2] = (int)(64u << 16);    // tensor_dim1 = 64 rows
            g1[3] = (int)(512u << 16);   // tile_dim0 = 512
            g1[4] = 64;                  // tile_dim1 = 64
            g1[5] = 512;                 // tensor_dim0_stride = 512 elements
            g1[6] = 0;
            g1[7] = 0;
            i32x4 gz4 = {0, 0, 0, 0};
            i32x8 gz8 = {0, 0, 0, 0, 0, 0, 0, 0};
            __builtin_amdgcn_tensor_load_to_lds(g0, g1, gz4, gz4, gz8, 0);
            __builtin_amdgcn_s_wait_tensorcnt(0);
        }
#else
        // ---- stage K tile (row-major, 64 x 512 bf16) via vmem ----
#pragma unroll
        for (int i = 0; i < 32; ++i) {
            int lin = i * 512 + tid * 4;
            int r = lin >> 8;        // 256 dwords per row
            int cdw = lin & 255;
            const u32x4* sp = (const u32x4*)(K32 + (headBase + t0 + r) * (DD / 2) + cdw);
            *(u32x4*)&Ks[r * KS_STRIDE + cdw] = *sp;
        }
#endif
        // ---- stage V tile transposed: Vs[d][t] ----
#pragma unroll
        for (int i = 0; i < 16; ++i) {
            int lin = i * 128 + tid;  // unit = 8 bf16
            int r = lin >> 6;         // 64 units per 512-wide row
            int u8 = lin & 63;
            u32x4 val = *(const u32x4*)(V32 + (headBase + t0 + r) * (DD / 2) + u8 * 4);
            const unsigned short* vsrc = (const unsigned short*)&val;
#pragma unroll
            for (int j = 0; j < 8; ++j)
                Vs16[(u8 * 8 + j) * (VS_STRIDE * 2) + r] = vsrc[j];
        }
        __syncthreads();

        // ---- S = Q K^T (16 rows x 64 cols per wave) ----
        v8f s[4];
#pragma unroll
        for (int j = 0; j < 4; ++j) s[j] = vzero;
#pragma unroll
        for (int ds = 0; ds < 16; ++ds) {
#pragma unroll
            for (int j = 0; j < 4; ++j) {
                FragBF kf;
                const u32x4* p =
                    (const u32x4*)(&Ks[(j * 16 + nn) * KS_STRIDE + ds * 16 + kh * 4]);
                kf.q[0] = p[0];
                kf.q[1] = p[2];
                s[j] = __builtin_amdgcn_wmma_f32_16x16x32_bf16(
                    false, qf[ds].v, false, kf.v, (short)0, s[j], false, false);
            }
        }

        // ---- online softmax (row groups are 16-lane halves; shfl masks <= 8) ----
        float al[8];
#pragma unroll
        for (int v = 0; v < 8; ++v) {
            float rm = -1e30f;
#pragma unroll
            for (int j = 0; j < 4; ++j) rm = fmaxf(rm, s[j][v] * SCALE);
#pragma unroll
            for (int mk = 8; mk >= 1; mk >>= 1) rm = fmaxf(rm, __shfl_xor(rm, mk, 32));
            float mnew = fmaxf(mprev[v], rm);
            float alpha = __expf(mprev[v] - mnew);
            float rs = 0.0f;
#pragma unroll
            for (int j = 0; j < 4; ++j) {
                float p = __expf(s[j][v] * SCALE - mnew);
                s[j][v] = p;
                rs += p;
            }
#pragma unroll
            for (int mk = 8; mk >= 1; mk >>= 1) rs += __shfl_xor(rs, mk, 32);
            lsum[v] = lsum[v] * alpha + rs;
            mprev[v] = mnew;
            al[v] = alpha;
        }
#pragma unroll
        for (int n = 0; n < 32; ++n)
#pragma unroll
            for (int v = 0; v < 8; ++v) o[n][v] *= al[v];

        // ---- write P (C layout) to LDS, re-read as A layout ----
#pragma unroll
        for (int j = 0; j < 4; ++j)
#pragma unroll
            for (int v = 0; v < 8; ++v)
                Ps16[(wave * 16 + v + 8 * kh) * (PS_STRIDE * 2) + j * 16 + nn] =
                    f2bf(s[j][v]);

        // ---- O += P V ----
#pragma unroll
        for (int ts = 0; ts < 2; ++ts) {
            FragBF pf;
            {
                int prow = wave * 16 + nn;
                const u32x2* p0 = (const u32x2*)(&Ps[prow * PS_STRIDE + ts * 16 + kh * 4]);
                const u32x2* p1 =
                    (const u32x2*)(&Ps[prow * PS_STRIDE + ts * 16 + 8 + kh * 4]);
                pf.d2[0] = p0[0];
                pf.d2[1] = p0[1];
                pf.d2[2] = p1[0];
                pf.d2[3] = p1[1];
            }
#pragma unroll
            for (int n = 0; n < 32; ++n) {
                FragBF vf;
#pragma unroll
                for (int i = 0; i < 8; ++i)
                    vf.u[i] = Vs[(n * 16 + nn) * VS_STRIDE + ts * 16 + kh * 4 + (i & 3) + (i >> 2) * 8];
                o[n] = __builtin_amdgcn_wmma_f32_16x16x32_bf16(
                    false, pf.v, false, vf.v, (short)0, o[n], false, false);
            }
        }
        __syncthreads();
    }

    // ---- finalize: divide by row sum, store fp32 per-head output ----
#pragma unroll
    for (int v = 0; v < 8; ++v) {
        float inv = 1.0f / lsum[v];
        size_t row = headBase + q0 + wave * 16 + v + 8 * kh;
        float* orow = O + row * DD;
#pragma unroll
        for (int n = 0; n < 32; ++n) orow[n * 16 + nn] = o[n][v] * inv;
    }
}

// ---------------------------------------------------------------------------
// Host-side launch
// ---------------------------------------------------------------------------
extern "C" void kernel_launch(void* const* d_in, const int* in_sizes, int n_in,
                              void* d_out, int out_size, void* d_ws, size_t ws_size,
                              hipStream_t stream) {
    (void)in_sizes; (void)n_in; (void)out_size; (void)ws_size;
    const float* x   = (const float*)d_in[0];
    const float* g1  = (const float*)d_in[1];
    const float* be1 = (const float*)d_in[2];
    const float* Wq  = (const float*)d_in[3];
    const float* bq  = (const float*)d_in[4];
    const float* Wk  = (const float*)d_in[5];
    const float* bk  = (const float*)d_in[6];
    const float* Wv  = (const float*)d_in[7];
    const float* bv  = (const float*)d_in[8];
    const float* Wo  = (const float*)d_in[9];
    const float* bo  = (const float*)d_in[10];
    const float* g2  = (const float*)d_in[11];
    const float* be2 = (const float*)d_in[12];
    const float* W1  = (const float*)d_in[13];
    const float* b1  = (const float*)d_in[14];
    const float* W2  = (const float*)d_in[15];
    const float* b2  = (const float*)d_in[16];
    float* out = (float*)d_out;

    char* ws = (char*)d_ws;
    size_t off = 0;
    auto alloc = [&](size_t bytes) -> char* {
        char* p = ws + off;
        off = (off + bytes + 255) & ~(size_t)255;
        return p;
    };

    const long long WQKV = (long long)HH * DD * DD;
    unsigned short* wq_bf = (unsigned short*)alloc((size_t)WQKV * 2);
    unsigned short* wk_bf = (unsigned short*)alloc((size_t)WQKV * 2);
    unsigned short* wv_bf = (unsigned short*)alloc((size_t)WQKV * 2);
    unsigned short* wo_bf = (unsigned short*)alloc((size_t)DD * DD * 2);
    unsigned short* w1_bf = (unsigned short*)alloc((size_t)DD * FF * 2);
    unsigned short* w2_bf = (unsigned short*)alloc((size_t)FF * DD * 2);
    unsigned short* h1_bf = (unsigned short*)alloc((size_t)MM * DD * 2);
    unsigned short* q_bf  = (unsigned short*)alloc((size_t)HH * MM * DD * 2);
    unsigned short* k_bf  = (unsigned short*)alloc((size_t)HH * MM * DD * 2);
    unsigned short* v_bf  = (unsigned short*)alloc((size_t)HH * MM * DD * 2);
    float*          Obuf  = (float*)alloc((size_t)HH * MM * DD * 4);
    unsigned short* avg_bf= (unsigned short*)alloc((size_t)MM * DD * 2);
    float*          x2    = (float*)alloc((size_t)MM * DD * 4);
    unsigned short* h2_bf = (unsigned short*)alloc((size_t)MM * DD * 2);
    unsigned short* m_bf  = (unsigned short*)alloc((size_t)MM * FF * 2);

    // 1) weights -> bf16
    auto conv = [&](const float* src, unsigned short* dst, long long n) {
        f32_to_bf16_kernel<<<dim3((unsigned)((n + 255) / 256)), 256, 0, stream>>>(src, dst, n);
    };
    conv(Wq, wq_bf, WQKV);
    conv(Wk, wk_bf, WQKV);
    conv(Wv, wv_bf, WQKV);
    conv(Wo, wo_bf, (long long)DD * DD);
    conv(W1, w1_bf, (long long)DD * FF);
    conv(W2, w2_bf, (long long)FF * DD);

    // 2) LN1
    layernorm_bf16_kernel<<<MM, 256, 0, stream>>>(x, g1, be1, h1_bf);

    // 3) QKV projections (grid.z = heads)
    dim3 gqkv(DD / 128, MM / 128, HH);
    gemm_bf16_kernel<0><<<gqkv, 128, 0, stream>>>(
        h1_bf, wq_bf, bq, nullptr, q_bf, nullptr, MM, DD, DD,
        (long long)DD * DD, (long long)DD, (long long)MM * DD);
    gemm_bf16_kernel<0><<<gqkv, 128, 0, stream>>>(
        h1_bf, wk_bf, bk, nullptr, k_bf, nullptr, MM, DD, DD,
        (long long)DD * DD, (long long)DD, (long long)MM * DD);
    gemm_bf16_kernel<0><<<gqkv, 128, 0, stream>>>(
        h1_bf, wv_bf, bv, nullptr, v_bf, nullptr, MM, DD, DD,
        (long long)DD * DD, (long long)DD, (long long)MM * DD);

    // 4) flash attention per (h,b, 64-row Q block)
    size_t flash_smem = (size_t)FLASH_LDS_DW * 4;
    flash_attn_kernel<<<dim3(SS / 64, HH * BB), 128, flash_smem, stream>>>(
        q_bf, k_bf, v_bf, Obuf);

    // 5) head mean
    long long md = (long long)MM * DD;
    avg_heads_kernel<<<dim3((unsigned)((md + 255) / 256)), 256, 0, stream>>>(Obuf, avg_bf, md);

    // 6) Wo projection + bias + residual(x) -> x2 (fp32)
    gemm_bf16_kernel<1><<<dim3(DD / 128, MM / 128, 1), 128, 0, stream>>>(
        avg_bf, wo_bf, bo, x, nullptr, x2, MM, DD, DD, 0, 0, 0);

    // 7) LN2
    layernorm_bf16_kernel<<<MM, 256, 0, stream>>>(x2, g2, be2, h2_bf);

    // 8) W1 + bias + GELU -> m (bf16)
    gemm_bf16_kernel<2><<<dim3(FF / 128, MM / 128, 1), 128, 0, stream>>>(
        h2_bf, w1_bf, b1, nullptr, m_bf, nullptr, MM, FF, DD, 0, 0, 0);

    // 9) W2 + bias + residual(x2) -> out (fp32)
    gemm_bf16_kernel<1><<<dim3(DD / 128, MM / 128, 1), 128, 0, stream>>>(
        m_bf, w2_bf, b2, x2, nullptr, out, MM, DD, FF, 0, 0, 0);
}
